// LSTM_74122545594502
// MI455X (gfx1250) — matrix operand order repaired and checked by reading, not compiled, not run on hardware
//
#include <hip/hip_runtime.h>
#include <hip/hip_bf16.h>

// ---------------------------------------------------------------------------
// Bidirectional 3-layer LSTM (B=4096, T=256, D=4, H=16) on gfx1250.
//
// Per wave: 16 batch rows. Per cell per timestep: one v_wmma_f32_16x16x32_f16
// per 16-gate tile (4 tiles = i,f,g,o), f16 A/B, f32 accum with C=0 (inline),
// biases folded into the activation transcendentals (prescaled by -log2 e so
// the existing exp-argument mul becomes an fma -- zero extra cost, and no
// 8-register broadcast-bias C operands to hoist).
// h state lives persistently in A-matrix register layout (4 packed-f16 VGPRs
// per cell); C-layout -> A-layout transpose goes through a small LDS stage.
// Weights live in LDS as f16 (shared by all 8 waves), fetched as ds_load_b128
// each iteration (address laundered so 192 VGPRs of weights don't get LICM'd).
// ---------------------------------------------------------------------------

typedef __attribute__((ext_vector_type(16))) _Float16 v16h;
typedef __attribute__((ext_vector_type(8)))  float    v8f;
typedef __attribute__((ext_vector_type(2)))  _Float16 v2h;

union AB {
    v16h v;
    v2h  p[8];
    int4 q[2];
};

#define NCELL   6
#define ROWPAD  40            // halves per weight row (32 used, padded to 80B)
#define CELLW   (64 * ROWPAD) // halves per cell weight block
#define STRIDE  18            // floats per batch row in transpose stage
#define TT      256
#define LOG2E_F 1.4426950408889634f

struct Params {
    const float* y;
    const float* Wih[6];
    const float* Whh[6];
    const float* bih[6];
    const float* bhh[6];
    const float* Wout;
    const float* bout;
    float*       out;
};

#if defined(__has_builtin)
#if __has_builtin(__builtin_amdgcn_exp2f)
#define EXP2(x) __builtin_amdgcn_exp2f(x)
#endif
#if __has_builtin(__builtin_amdgcn_tanhf)
#define ATH_HAVE_TANH 1
#endif
#endif
#ifndef EXP2
#define EXP2(x) exp2f(x)
#endif

__device__ __forceinline__ v2h pkh(float a, float b) {
    return __builtin_bit_cast(v2h, __builtin_amdgcn_cvt_pkrtz(a, b));
}

// sigmoid(x + b) with bpre = -log2(e) * b  (bias rides the fma for free)
__device__ __forceinline__ float sigm_pre(float x, float bpre) {
    float e = EXP2(__builtin_fmaf(x, -LOG2E_F, bpre));
    return __builtin_amdgcn_rcpf(1.0f + e);
}

#ifdef ATH_HAVE_TANH
#define GBIAS(b) (b)   // raw bias; one v_add before v_tanh_f32
__device__ __forceinline__ float tanh_pre(float x, float b) {
    return __builtin_amdgcn_tanhf(x + b);
}
__device__ __forceinline__ float ftanh(float x) {
    return __builtin_amdgcn_tanhf(x);
}
#else
#define GBIAS(b) ((b) * (-2.0f * LOG2E_F))   // prescaled for the fma
__device__ __forceinline__ float tanh_pre(float x, float bpre2) {
    float e = EXP2(__builtin_fmaf(x, -2.0f * LOG2E_F, bpre2));
    return (1.0f - e) * __builtin_amdgcn_rcpf(1.0f + e);
}
__device__ __forceinline__ float ftanh(float x) {
    float e = EXP2(-2.0f * LOG2E_F * x);
    return (1.0f - e) * __builtin_amdgcn_rcpf(1.0f + e);
}
#endif

// Assemble A (16x32 f16): K0..15 = lo pairs, K16..31 = hi pairs (per-lane).
__device__ __forceinline__ v16h packA(const v2h* lo, const v2h* hi) {
    AB a;
#pragma unroll
    for (int i = 0; i < 4; ++i) { a.p[i] = lo[i]; a.p[4 + i] = hi[i]; }
    return a.v;
}

// One LSTM cell step for a 16-batch tile.
//  wcell : LDS f16 weights, 64 rows (gates) x ROWPAD halves; B[k][gate]
//  bias4 : per-lane gate biases (i,f prescaled -log2e; g per GBIAS; o -log2e)
//  cs    : persistent cell state, C layout (batch = r + 8*half, hid = n)
//  hv    : persistent h state in A layout (4 packed pairs), overwritten
__device__ __forceinline__ void lstm_cell(const _Float16* __restrict__ wcell,
                                          const float* __restrict__ bias4,
                                          v16h A, v8f& cs, v2h* __restrict__ hv,
                                          float* __restrict__ stage,
                                          int n, int half) {
    v8f acc[4];
    const v8f zc = {};
#pragma unroll
    for (int tile = 0; tile < 4; ++tile) {
        const _Float16* wt = wcell + ((tile * 16 + n) * ROWPAD + 8 * half);
        AB B;
        B.q[0] = *(const int4*)(wt);       // ds_load_b128: K = 8h + 0..7
        B.q[1] = *(const int4*)(wt + 16);  // ds_load_b128: K = 16 + 8h + 0..7
        acc[tile] = __builtin_amdgcn_wmma_f32_16x16x32_f16(
            false, A, false, B.v, (short)0, zc, false, false);
    }
    // Activations + state update in C layout, stage h to LDS (batch-major).
#pragma unroll
    for (int r = 0; r < 8; ++r) {
        float ig = sigm_pre(acc[0][r], bias4[0]);
        float fg = sigm_pre(acc[1][r], bias4[1]);
        float gg = tanh_pre(acc[2][r], bias4[2]);
        float og = sigm_pre(acc[3][r], bias4[3]);
        float cn = fg * cs[r] + ig * gg;
        cs[r]    = cn;
        stage[(r + 8 * half) * STRIDE + n] = og * ftanh(cn);
    }
    // Transpose read-back: lane = batch n, pairs of hid (8h+2p, 8h+2p+1).
#pragma unroll
    for (int p = 0; p < 4; ++p) {
        const float2 tv = *(const float2*)(stage + (n * STRIDE + 8 * half + 2 * p));
        hv[p] = pkh(tv.x, tv.y);
    }
}

__global__ __launch_bounds__(256) void lstm_bidir_kernel(Params P) {
    __shared__ _Float16 wfs[NCELL * CELLW];       // 30720 B
    __shared__ float    bsum[NCELL * 64];         //  1536 B
    __shared__ float    stage[8][2][16 * STRIDE]; // 18432 B

    const int tid = threadIdx.x;

    // --- Build fused f16 weight matrix in LDS: B[k][gate] per cell. ---
    // layer-1 cells (c=0 fwd, c=3 bwd):  K0..15 = Whh (own h), K16..19 = Wih (x)
    // layer-2/3 cells:                   K0..15 = Wih (h_in),  K16..31 = Whh (own h)
    for (int row = tid; row < NCELL * 64; row += 256) {
        int c = row >> 6, g = row & 63;
        bool l1 = (c == 0 || c == 3);
        const float* Wih = P.Wih[c];
        const float* Whh = P.Whh[c];
        _Float16* wr = &wfs[row * ROWPAD];
#pragma unroll
        for (int k = 0; k < ROWPAD; ++k) {
            float w = 0.0f;
            if (l1) {
                if (k < 16)      w = Whh[g * 16 + k];
                else if (k < 20) w = Wih[g * 4 + (k - 16)];
            } else {
                if (k < 16)      w = Wih[g * 16 + k];
                else if (k < 32) w = Whh[g * 16 + (k - 16)];
            }
            wr[k] = (_Float16)w;
        }
        bsum[row] = P.bih[c][g] + P.bhh[c][g];
    }
    __syncthreads();

    const int wave = tid >> 5, lane = tid & 31;
    const int n = lane & 15, half = lane >> 4;
    const int batch = blockIdx.x * 128 + wave * 16 + n;

    // Per-lane gate biases, prescaled so they fuse into the exp-argument fma.
    float biasReg[24];
#pragma unroll
    for (int c = 0; c < NCELL; ++c)
#pragma unroll
        for (int tl = 0; tl < 4; ++tl) {
            float bv = bsum[c * 64 + tl * 16 + n];
            biasReg[c * 4 + tl] = (tl == 2) ? GBIAS(bv) : bv * (-LOG2E_F);
        }

    v2h Hv[NCELL][4];
    v8f cs[NCELL];
    {
        v2h z2 = {};
        v8f zf = {};
#pragma unroll
        for (int c = 0; c < NCELL; ++c) {
            cs[c] = zf;
#pragma unroll
            for (int p = 0; p < 4; ++p) Hv[c][p] = z2;
        }
    }

    const float* __restrict__ yb = P.y + (size_t)batch * (TT * 4);
    float* stf = stage[wave][0];
    float* stb = stage[wave][1];

#pragma unroll 1
    for (int t = 0; t < TT; ++t) {
        // Launder LDS base each iteration so B tiles reload from LDS instead
        // of being LICM-hoisted into ~192 VGPRs (which would spill).
        unsigned lz = 0;
        asm volatile("" : "+v"(lz));
        const _Float16* wb = wfs + lz;

        float4 xf = *(const float4*)(yb + t * 4);              // fwd input
        float4 xb = *(const float4*)(yb + (TT - 1 - t) * 4);   // bwd input
        __builtin_prefetch(yb + (t + 1) * 4, 0, 3);            // global_prefetch_b8

        v2h z2 = {};
        // Interleave the two independent chains: f1,b1,f2,b2,f3,b3.
        {
            v2h xh[4];
            xh[0] = half ? z2 : pkh(xf.x, xf.y);
            xh[1] = half ? z2 : pkh(xf.z, xf.w);
            xh[2] = z2; xh[3] = z2;
            lstm_cell(wb + 0 * CELLW, biasReg + 0, packA(Hv[0], xh),
                      cs[0], Hv[0], stf, n, half);
        }
        {
            v2h xh[4];
            xh[0] = half ? z2 : pkh(xb.x, xb.y);
            xh[1] = half ? z2 : pkh(xb.z, xb.w);
            xh[2] = z2; xh[3] = z2;
            lstm_cell(wb + 3 * CELLW, biasReg + 12, packA(Hv[3], xh),
                      cs[3], Hv[3], stb, n, half);
        }
        lstm_cell(wb + 1 * CELLW, biasReg + 4, packA(Hv[0], Hv[1]),
                  cs[1], Hv[1], stf, n, half);
        lstm_cell(wb + 4 * CELLW, biasReg + 16, packA(Hv[3], Hv[4]),
                  cs[4], Hv[4], stb, n, half);
        lstm_cell(wb + 2 * CELLW, biasReg + 8, packA(Hv[1], Hv[2]),
                  cs[2], Hv[2], stf, n, half);
        lstm_cell(wb + 5 * CELLW, biasReg + 20, packA(Hv[4], Hv[5]),
                  cs[5], Hv[5], stb, n, half);
    }

    // ---- output projection: out[b] = [hf3 | hb3] @ Wout^T + bout ----
    float pr[4] = {0.f, 0.f, 0.f, 0.f};
#pragma unroll
    for (int q = 0; q < 4; ++q)
#pragma unroll
        for (int s = 0; s < 2; ++s) {
            int   k  = 8 * half + 2 * q + s;
            float hf = (float)Hv[2][q][s];
            float hb = (float)Hv[5][q][s];
#pragma unroll
            for (int j = 0; j < 4; ++j)
                pr[j] += hf * P.Wout[j * 32 + k] + hb * P.Wout[j * 32 + 16 + k];
        }
    float* ex = stf;  // reuse stage buffer; in-wave LDS ops are in order
#pragma unroll
    for (int j = 0; j < 4; ++j) ex[(half * 16 + n) * 4 + j] = pr[j];
    if (half == 0) {
        float4 r;
        r.x = ex[n * 4 + 0] + ex[(16 + n) * 4 + 0] + P.bout[0];
        r.y = ex[n * 4 + 1] + ex[(16 + n) * 4 + 1] + P.bout[1];
        r.z = ex[n * 4 + 2] + ex[(16 + n) * 4 + 2] + P.bout[2];
        r.w = ex[n * 4 + 3] + ex[(16 + n) * 4 + 3] + P.bout[3];
        *(float4*)(P.out + (size_t)batch * 4) = r;
    }
}

extern "C" void kernel_launch(void* const* d_in, const int* in_sizes, int n_in,
                              void* d_out, int out_size, void* d_ws, size_t ws_size,
                              hipStream_t stream) {
    (void)n_in; (void)out_size; (void)d_ws; (void)ws_size;
    Params p;
    p.y = (const float*)d_in[0];
    for (int c = 0; c < 6; ++c) {           // order: f1,f2,f3,b1,b2,b3
        p.Wih[c] = (const float*)d_in[1 + c * 4 + 0];
        p.Whh[c] = (const float*)d_in[1 + c * 4 + 1];
        p.bih[c] = (const float*)d_in[1 + c * 4 + 2];
        p.bhh[c] = (const float*)d_in[1 + c * 4 + 3];
    }
    p.Wout = (const float*)d_in[25];
    p.bout = (const float*)d_in[26];
    p.out  = (float*)d_out;

    const int B = in_sizes[0] / (TT * 4);   // 4096
    const int grid = B / 128;               // 16 batches/wave * 8 waves/block
    lstm_bidir_kernel<<<grid, 256, 0, stream>>>(p);
}
